// simpleHGN_82952998355817
// MI455X (gfx1250) — matrix operand order repaired and checked by reading, not compiled, not run on hardware
//
#include <hip/hip_runtime.h>
#include <hip/hip_bf16.h>

// ---------------- problem constants ----------------
#define NN   50000
#define EE   800000
#define TT   8
#define INF_ 256
#define HID_ 64
#define H1_  4
#define OUT_ 64
#define H2_  1
#define ED_  256

typedef _Float16 v16h  __attribute__((ext_vector_type(16)));
typedef _Float16 half8 __attribute__((ext_vector_type(8)));
typedef float    v8f   __attribute__((ext_vector_type(8)));

// ---------------- helpers ----------------
__device__ __forceinline__ unsigned f32_ord(float x) {
    unsigned u = __float_as_uint(x);
    return (u & 0x80000000u) ? ~u : (u | 0x80000000u);
}
__device__ __forceinline__ float ord_f32(unsigned u) {
    return __uint_as_float((u & 0x80000000u) ? (u & 0x7fffffffu) : ~u);
}

// ---------------- f32 -> f16 convert ----------------
__global__ void cvt_f32_f16(const float* __restrict__ src,
                            _Float16* __restrict__ dst, int n) {
    int i = blockIdx.x * blockDim.x + threadIdx.x;
    if (i < n) dst[i] = (_Float16)src[i];
}

// ---------------- f32[K][Nc] -> f16 transposed [Nc][K] ----------------
__global__ void cvt_transpose_f16(const float* __restrict__ src,
                                  _Float16* __restrict__ dst, int K, int Nc) {
    int i = blockIdx.x * blockDim.x + threadIdx.x;     // coalesced read of src
    if (i >= K * Nc) return;
    int k = i / Nc, n = i - k * Nc;
    dst[(size_t)n * K + k] = (_Float16)src[i];
}

// ---------------- WMMA GEMM: C[M,Nc] = A[M,K] * B[K,Nc] (+bias) ----------------
// B supplied TRANSPOSED (Bt[Nc][K]) so both fragments load as contiguous b128s.
// Each wave computes a 16x64 output block: A fragment reused across 4 WMMAs.
template<int KDIM>
__global__ void gemm_wmma_f16(const _Float16* __restrict__ A,
                              const _Float16* __restrict__ Bt,
                              const float* __restrict__ bias,
                              float* __restrict__ C, int M, int Nc) {
    const int lane = threadIdx.x & 31;
    const int wave = threadIdx.x >> 5;
    const int nb   = Nc >> 6;                          // number of 64-col blocks
    const int tile = blockIdx.x * (blockDim.x >> 5) + wave;
    if (tile >= (M >> 4) * nb) return;                 // wave-uniform exit
    const int tm  = tile / nb;
    const int tn4 = tile - tm * nb;
    const int g = lane >> 4;                           // half-wave group
    const int l = lane & 15;

    v8f c[4] = {{}, {}, {}, {}};
    const _Float16* Arow = A  + (size_t)(tm * 16 + l) * KDIM;
    const _Float16* Bc0  = Bt + (size_t)(tn4 * 64 + l) * KDIM;

#pragma unroll
    for (int k0 = 0; k0 < KDIM; k0 += 32) {
        // A: halves 0..7 -> K = k0+g*8+i ; halves 8..15 -> K = k0+16+g*8+(i-8)
        half8 alo = *(const half8*)(Arow + k0 + g * 8);
        half8 ahi = *(const half8*)(Arow + k0 + 16 + g * 8);
        v16h a;
#pragma unroll
        for (int i = 0; i < 8; ++i) { a[i] = alo[i]; a[8 + i] = ahi[i]; }

#pragma unroll
        for (int j = 0; j < 4; ++j) {
            // B (transposed): lane = column, halves 0..7 -> K = k0+g*8+i ; halves 8..15 -> K = k0+16+g*8+(i-8)
            const _Float16* Bp = Bc0 + (size_t)j * 16 * KDIM + k0 + g * 8;
            half8 blo = *(const half8*)(Bp);
            half8 bhi = *(const half8*)(Bp + 16);
            v16h b;
#pragma unroll
            for (int i = 0; i < 8; ++i) { b[i] = blo[i]; b[8 + i] = bhi[i]; }
            c[j] = __builtin_amdgcn_wmma_f32_16x16x32_f16(
                       false, a, false, b, (short)0, c[j], false, false);
        }
    }
#pragma unroll
    for (int j = 0; j < 4; ++j) {
        const int ncol = tn4 * 64 + j * 16 + l;
        const float bv = bias ? bias[ncol] : 0.0f;
#pragma unroll
        for (int r = 0; r < 8; ++r) {                  // C row: M = tm*16 + r + 8*g
            C[(size_t)(tm * 16 + r + 8 * g) * Nc + ncol] = c[j][r] + bv;
        }
    }
}

// ---------------- node attention scores: hl/hr[n,h] = sum_d a*emb ----------------
template<int H, int D>
__global__ void node_scores(const float* __restrict__ emb,
                            const float* __restrict__ a_l,
                            const float* __restrict__ a_r,
                            float* __restrict__ hl, float* __restrict__ hr, int n) {
    int idx = blockIdx.x * blockDim.x + threadIdx.x;
    if (idx >= n * H) return;
    int node = idx / H, h = idx & (H - 1);
    const float* e  = emb + (size_t)node * H * D + h * D;
    const float* al = a_l + h * D;
    const float* ar = a_r + h * D;
    float sl = 0.f, sr = 0.f;
#pragma unroll 8
    for (int i = 0; i < D; ++i) { float v = e[i]; sl += al[i] * v; sr += ar[i] * v; }
    hl[idx] = sl; hr[idx] = sr;
}

// ---------------- per-type edge scores: het[t,h] ----------------
// blockDim must be ED_ (=256). block = (t,h).
template<int H>
__global__ void etype_scores(const float* __restrict__ edge_emb,
                             const float* __restrict__ Wr,
                             const float* __restrict__ a_e,
                             float* __restrict__ het) {
    const int t = blockIdx.x / H;
    const int h = blockIdx.x - t * H;
    const int e = threadIdx.x;
    const float* emb = edge_emb + (size_t)t * ED_;
    const float* W   = Wr + (size_t)t * ED_ * ED_ * H + (size_t)h * ED_ + e;
    float acc = 0.f;
    for (int k = 0; k < ED_; ++k) acc += emb[k] * W[(size_t)k * ED_ * H];
    acc *= a_e[(size_t)h * ED_ + e];
    __shared__ float red[ED_];
    red[e] = acc; __syncthreads();
    for (int s = ED_ / 2; s > 0; s >>= 1) {
        if (e < s) red[e] += red[e + s];
        __syncthreads();
    }
    if (e == 0) het[t * H + h] = red[0];
}

// ---------------- edge attention + segment max ----------------
template<int H>
__global__ void att_max(const float* __restrict__ hl, const float* __restrict__ hr,
                        const float* __restrict__ het,
                        const int* __restrict__ row, const int* __restrict__ col,
                        const int* __restrict__ et,
                        float* __restrict__ att, unsigned* __restrict__ amax) {
    int idx = blockIdx.x * blockDim.x + threadIdx.x;
    if (idx >= EE * H) return;
    int e = idx / H, h = idx & (H - 1);
    float v = hl[row[e] * H + h] + het[et[e] * H + h] + hr[col[e] * H + h];
    v = (v > 0.f) ? v : 0.2f * v;                      // leaky relu
    att[idx] = v;
    atomicMax(&amax[col[e] * H + h], f32_ord(v));
}

// ---------------- exp + segment sum ----------------
template<int H>
__global__ void exp_sum(const int* __restrict__ col,
                        float* __restrict__ att, const unsigned* __restrict__ amax,
                        float* __restrict__ denom) {
    int idx = blockIdx.x * blockDim.x + threadIdx.x;
    if (idx >= EE * H) return;
    int e = idx / H, h = idx & (H - 1);
    float m  = ord_f32(amax[col[e] * H + h]);
    float ex = expf(att[idx] - m);
    att[idx] = ex;
    atomicAdd(&denom[col[e] * H + h], ex);
}

// ---------------- weighted message scatter ----------------
// out[col, dd*H + h] += emb[row, h*D + dd] * alpha[e,h]
template<int H, int D>
__global__ void scatter_msg(const float* __restrict__ emb, const float* __restrict__ ex,
                            const float* __restrict__ denom,
                            const int* __restrict__ row, const int* __restrict__ col,
                            float* __restrict__ out) {
    constexpr int F = H * D;
    int gid = blockIdx.x * blockDim.x + threadIdx.x;
    if (gid >= EE * F) return;
    int e  = gid / F;
    int j  = gid & (F - 1);                            // emb index: j = h*D + dd
    int h  = j / D, dd = j & (D - 1);
    int c  = col[e];
    float alpha = ex[(size_t)e * H + h] / denom[(size_t)c * H + h];
    float v = emb[(size_t)row[e] * F + j] * alpha;
    atomicAdd(&out[(size_t)c * F + dd * H + h], v);
}

// ---------------- elu in-place + convert to f16 ----------------
__global__ void elu_cvt(float* __restrict__ x, _Float16* __restrict__ xf, int n) {
    int i = blockIdx.x * blockDim.x + threadIdx.x;
    if (i >= n) return;
    float v = x[i];
    v = (v > 0.f) ? v : (expf(v) - 1.0f);
    x[i] = v;
    xf[i] = (_Float16)v;
}

// ==================================================================
extern "C" void kernel_launch(void* const* d_in, const int* in_sizes, int n_in,
                              void* d_out, int out_size, void* d_ws, size_t ws_size,
                              hipStream_t stream) {
    (void)in_sizes; (void)n_in; (void)out_size; (void)ws_size;
    const float* h     = (const float*)d_in[0];
    const int*   row   = (const int*)  d_in[1];
    const int*   col   = (const int*)  d_in[2];
    const int*   etype = (const int*)  d_in[3];
    const float* W1    = (const float*)d_in[4];
    const float* a_l1  = (const float*)d_in[5];
    const float* a_r1  = (const float*)d_in[6];
    const float* a_e1  = (const float*)d_in[7];
    const float* eemb1 = (const float*)d_in[8];
    const float* Wr1   = (const float*)d_in[9];
    const float* W2    = (const float*)d_in[10];
    const float* a_l2  = (const float*)d_in[11];
    const float* a_r2  = (const float*)d_in[12];
    const float* a_e2  = (const float*)d_in[13];
    const float* eemb2 = (const float*)d_in[14];
    const float* Wr2   = (const float*)d_in[15];
    const float* resW  = (const float*)d_in[16];
    const float* resb  = (const float*)d_in[17];
    float* out = (float*)d_out;

    // ---- workspace carve (256B aligned) ----
    char* ws = (char*)d_ws;
    size_t off = 0;
    auto carve = [&](size_t bytes) -> char* {
        char* p = ws + off;
        off += (bytes + 255) & ~(size_t)255;
        return p;
    };
    const int F1 = HID_ * H1_;                         // 256
    const int F2 = OUT_ * H2_;                         // 64
    _Float16* hf16  = (_Float16*)carve((size_t)NN * INF_ * 2);  // reused as h1f16
    _Float16* w1t   = (_Float16*)carve((size_t)INF_ * F1 * 2);  // transposed f16 W1
    float*    emb1  = (float*)   carve((size_t)NN * F1 * 4);    // reused as emb2
    float*    hl    = (float*)   carve((size_t)NN * H1_ * 4);
    float*    hr    = (float*)   carve((size_t)NN * H1_ * 4);
    float*    het   = (float*)   carve((size_t)TT * H1_ * 4);
    float*    att   = (float*)   carve((size_t)EE * H1_ * 4);
    unsigned* amax  = (unsigned*)carve((size_t)NN * H1_ * 4);
    float*    denom = (float*)   carve((size_t)NN * H1_ * 4);
    float*    h1    = (float*)   carve((size_t)NN * F1 * 4);
    _Float16* w2t   = (_Float16*)carve((size_t)F1 * F2 * 2);    // transposed f16 W2
    _Float16* rwt   = (_Float16*)carve((size_t)F1 * F2 * 2);    // transposed f16 resW
    _Float16* h1f16 = hf16;                            // reuse (h dead after gemm1)
    float*    emb2  = emb1;                            // reuse (emb1 dead after L1 scatter)

    const int TB = 256;
    auto nblk = [](long long n, int b) { return (int)((n + b - 1) / b); };

    // ================= Layer 1 =================
    cvt_f32_f16<<<nblk((long long)NN * INF_, TB), TB, 0, stream>>>(h, hf16, NN * INF_);
    cvt_transpose_f16<<<nblk((long long)INF_ * F1, TB), TB, 0, stream>>>(W1, w1t, INF_, F1);

    {   // emb1 = h @ W1 : M=50000, Nc=256, K=256 -> 16x64 block per wave
        int tiles = (NN / 16) * (F1 / 64);
        gemm_wmma_f16<INF_><<<nblk(tiles, 4), 128, 0, stream>>>(hf16, w1t, nullptr,
                                                                emb1, NN, F1);
    }
    node_scores<H1_, HID_><<<nblk((long long)NN * H1_, TB), TB, 0, stream>>>(
        emb1, a_l1, a_r1, hl, hr, NN);
    etype_scores<H1_><<<TT * H1_, ED_, 0, stream>>>(eemb1, Wr1, a_e1, het);

    hipMemsetAsync(amax,  0, (size_t)NN * H1_ * 4, stream);
    hipMemsetAsync(denom, 0, (size_t)NN * H1_ * 4, stream);
    hipMemsetAsync(h1,    0, (size_t)NN * F1  * 4, stream);

    att_max<H1_><<<nblk((long long)EE * H1_, TB), TB, 0, stream>>>(hl, hr, het, row, col,
                                                                   etype, att, amax);
    exp_sum<H1_><<<nblk((long long)EE * H1_, TB), TB, 0, stream>>>(col, att, amax, denom);
    scatter_msg<H1_, HID_><<<nblk((long long)EE * F1, TB), TB, 0, stream>>>(
        emb1, att, denom, row, col, h1);
    elu_cvt<<<nblk((long long)NN * F1, TB), TB, 0, stream>>>(h1, h1f16, NN * F1);

    // ================= Layer 2 =================
    cvt_transpose_f16<<<nblk((long long)F1 * F2, TB), TB, 0, stream>>>(W2,   w2t, F1, F2);
    cvt_transpose_f16<<<nblk((long long)F1 * F2, TB), TB, 0, stream>>>(resW, rwt, F1, F2);

    {   // emb2 = h1 @ W2 ; out = h1 @ resW + resb  (residual pre-loaded into d_out)
        int tiles = (NN / 16) * (F2 / 64);
        gemm_wmma_f16<F1><<<nblk(tiles, 4), 128, 0, stream>>>(h1f16, w2t, nullptr,
                                                              emb2, NN, F2);
        gemm_wmma_f16<F1><<<nblk(tiles, 4), 128, 0, stream>>>(h1f16, rwt, resb,
                                                              out, NN, F2);
    }
    node_scores<H2_, OUT_><<<nblk((long long)NN * H2_, TB), TB, 0, stream>>>(
        emb2, a_l2, a_r2, hl, hr, NN);
    etype_scores<H2_><<<TT * H2_, ED_, 0, stream>>>(eemb2, Wr2, a_e2, het);

    hipMemsetAsync(amax,  0, (size_t)NN * H2_ * 4, stream);
    hipMemsetAsync(denom, 0, (size_t)NN * H2_ * 4, stream);

    att_max<H2_><<<nblk((long long)EE * H2_, TB), TB, 0, stream>>>(hl, hr, het, row, col,
                                                                   etype, att, amax);
    exp_sum<H2_><<<nblk((long long)EE * H2_, TB), TB, 0, stream>>>(col, att, amax, denom);
    scatter_msg<H2_, OUT_><<<nblk((long long)EE * F2, TB), TB, 0, stream>>>(
        emb2, att, denom, row, col, out);
}
